// MultiheadAttention_with_jax_45973329936611
// MI455X (gfx1250) — matrix operand order repaired and checked by reading, not compiled, not run on hardware
//
#include <hip/hip_runtime.h>

#define TDIM 1024
#define NHEADS 16
#define HD 64
#define NTOK 4096  // B*T = 2*2048
#define TPB 2048   // tokens per batch

typedef __attribute__((ext_vector_type(16))) __bf16 bf16x16;
typedef __attribute__((ext_vector_type(8)))  __bf16 bf16x8;
typedef __attribute__((ext_vector_type(8)))  float  f32x8;

// ---------- fp32 -> bf16 (round to nearest even), bit math for portability ----
__device__ __forceinline__ __bf16 f2bf(float f) {
    unsigned u = __builtin_bit_cast(unsigned, f);
    unsigned r = u + 0x7fffu + ((u >> 16) & 1u);
    unsigned short h = (unsigned short)(r >> 16);
    return __builtin_bit_cast(__bf16, h);
}

// ---------- fragment loaders (row-major source, ld in elements) --------------
// A-matrix 16x32 bf16: lane L: row M=L&15, half=L>>4.
// elements 0..7  = K = 8*half .. 8*half+7
// elements 8..15 = K = 16+8*half .. 16+8*half+7
__device__ __forceinline__ bf16x16 load_frag_a(const __bf16* p, int ld, int lane) {
    int r = lane & 15, h = lane >> 4;
    const __bf16* q = p + (size_t)r * ld + h * 8;
    union { bf16x16 v16; bf16x8 v8[2]; } u;
    u.v8[0] = *(const bf16x8*)(q);
    u.v8[1] = *(const bf16x8*)(q + 16);
    return u.v16;
}
// B-matrix 32x16 bf16 (source row-major [col, K]):
// lane L: col N=L&15, half=L>>4. elements 0..7 = K = 8*half .. 8*half+7, 8..15 = K = 16+8*half .. 16+8*half+7
__device__ __forceinline__ bf16x16 load_frag_b(const __bf16* p, int ld, int lane) {
    int c = lane & 15, h = lane >> 4;
    const __bf16* q = p + (size_t)c * ld + h * 8;
    union { bf16x16 v16; bf16x8 v8[2]; } u;
    u.v8[0] = *(const bf16x8*)(q);
    u.v8[1] = *(const bf16x8*)(q + 16);
    return u.v16;
}

#define WMMA_BF16(a, b, c) \
    __builtin_amdgcn_wmma_f32_16x16x32_bf16(false, (a), false, (b), (short)0, (c), false, false)

// ---------- fp32 -> bf16 conversion kernel -----------------------------------
__global__ void cvt_f32_bf16(const float* __restrict__ in, __bf16* __restrict__ out, int n) {
    int i = blockIdx.x * blockDim.x + threadIdx.x;
    int stride = gridDim.x * blockDim.x;
    for (; i < n; i += stride) out[i] = f2bf(in[i]);
}

// ---------- GEMM: Y = (A @ W^T + bias) * scale -------------------------------
// A: [M,K] bf16 row-major, W: [N,K] bf16 row-major.
// MODE 0: bf16 out [M,N].  MODE 1: bf16 out transposed [N, ldT] (Y^T).
// MODE 2: fp32 out [M,N].
// wave tile 64x64, 8 waves/block -> block tile 128x256.
template <int MODE>
__global__ void gemm_wmma(const __bf16* __restrict__ A, const __bf16* __restrict__ W,
                          const float* __restrict__ bias, void* __restrict__ outp,
                          int M, int N, int K, int ldT, float scale) {
    const int lane = threadIdx.x & 31;
    const int wave = threadIdx.x >> 5;
    const int m0 = blockIdx.x * 128 + (wave & 1) * 64;
    const int n0 = blockIdx.y * 256 + (wave >> 1) * 64;

    f32x8 acc[4][4] = {};
    for (int k0 = 0; k0 < K; k0 += 32) {
        bf16x16 af[4], bfr[4];
#pragma unroll
        for (int i = 0; i < 4; ++i)
            af[i] = load_frag_a(A + (size_t)(m0 + 16 * i) * K + k0, K, lane);
#pragma unroll
        for (int j = 0; j < 4; ++j)
            bfr[j] = load_frag_b(W + (size_t)(n0 + 16 * j) * K + k0, K, lane);
#pragma unroll
        for (int i = 0; i < 4; ++i)
#pragma unroll
            for (int j = 0; j < 4; ++j)
                acc[i][j] = WMMA_BF16(af[i], bfr[j], acc[i][j]);
    }

    const int c = lane & 15, h = lane >> 4;
#pragma unroll
    for (int i = 0; i < 4; ++i) {
#pragma unroll
        for (int j = 0; j < 4; ++j) {
            const int n = n0 + 16 * j + c;
            const float b = bias[n];
            if (MODE == 1) {
                bf16x8 o;
#pragma unroll
                for (int v = 0; v < 8; ++v) o[v] = f2bf((acc[i][j][v] + b) * scale);
                *(bf16x8*)((__bf16*)outp + (size_t)n * ldT + m0 + 16 * i + 8 * h) = o;
            } else {
#pragma unroll
                for (int v = 0; v < 8; ++v) {
                    const int m = m0 + 16 * i + v + 8 * h;
                    if (MODE == 0)
                        ((__bf16*)outp)[(size_t)m * N + n] = f2bf((acc[i][j][v] + b) * scale);
                    else
                        ((float*)outp)[(size_t)m * N + n] = (acc[i][j][v] + b) * scale;
                }
            }
        }
    }
}

// ---------- flash attention (transposed-S formulation) ------------------------
// Q,K: [NTOK, TDIM] bf16 (Q pre-scaled by 1/hd). Vt: [TDIM, NTOK] bf16 (V^T).
// One wave per (batch, head, 16 queries). Computes S^T = K_tile @ Q^T so each
// lane owns one query column: softmax reduces in-register + one xor-16 shuffle.
// Then O^T = Vt @ P^T via LDS re-layout of P (vectorized b128 LDS traffic).
__global__ void attn_flash(const __bf16* __restrict__ Q, const __bf16* __restrict__ Km,
                           const __bf16* __restrict__ Vt, __bf16* __restrict__ O) {
    __shared__ __bf16 lds[8][16 * 64];
    const int lane = threadIdx.x & 31;
    const int wave = threadIdx.x >> 5;
    const int gw = blockIdx.x * 8 + wave;    // 0..4095
    const int qt = gw & 127;                 // query tile within batch
    const int hh = (gw >> 7) & 15;           // head
    const int bb = gw >> 11;                 // batch
    const int m0 = bb * TPB + qt * 16;       // global token row of query tile
    const int nbase = bb * TPB;
    const int h = lane >> 4, c = lane & 15;

    // Q as B-operand (lane = query column), loaded once for the whole pass
    bf16x16 bq0 = load_frag_b(Q + (size_t)m0 * TDIM + hh * HD + 0, TDIM, lane);
    bf16x16 bq1 = load_frag_b(Q + (size_t)m0 * TDIM + hh * HD + 32, TDIM, lane);

    f32x8 oacc[4] = {};               // O^T accum: lane = query, rows = dims
    float mrun = -3.0e38f, lrun = 0.f;

    for (int kt = 0; kt < TPB / 64; ++kt) {
        const int n0 = nbase + kt * 64;
        // S^T tiles: rows = keys (4 x 16), cols = queries (16)
        f32x8 s[4] = {};
#pragma unroll
        for (int j = 0; j < 4; ++j) {
            bf16x16 a0 = load_frag_a(Km + (size_t)(n0 + 16 * j) * TDIM + hh * HD + 0, TDIM, lane);
            bf16x16 a1 = load_frag_a(Km + (size_t)(n0 + 16 * j) * TDIM + hh * HD + 32, TDIM, lane);
            s[j] = WMMA_BF16(a0, bq0, s[j]);
            s[j] = WMMA_BF16(a1, bq1, s[j]);
        }
        // online softmax over keys: this lane holds 32 of its query's 64 scores
        float mx = s[0][0];
#pragma unroll
        for (int j = 0; j < 4; ++j)
#pragma unroll
            for (int v = 0; v < 8; ++v) mx = fmaxf(mx, s[j][v]);
        mx = fmaxf(mx, __shfl_xor(mx, 16, 32));          // combine the two halves
        const float mnew = fmaxf(mrun, mx);
        const float corr = __expf(mrun - mnew);
        mrun = mnew;
        float psum = 0.f;
#pragma unroll
        for (int j = 0; j < 4; ++j)
#pragma unroll
            for (int v = 0; v < 8; ++v) {
                const float p = __expf(s[j][v] - mnew);
                s[j][v] = p;
                psum += p;
            }
        psum += __shfl_xor(psum, 16, 32);
        lrun = lrun * corr + psum;
#pragma unroll
        for (int j2 = 0; j2 < 4; ++j2) oacc[j2] *= corr;  // per-lane scalar splat

        // stage P row-major [query][key] 16x64 in per-wave LDS (vector stores:
        // this lane's 8 key values per tile are contiguous)
#pragma unroll
        for (int j = 0; j < 4; ++j) {
            bf16x8 o;
#pragma unroll
            for (int v = 0; v < 8; ++v) o[v] = f2bf(s[j][v]);
            *(bf16x8*)&lds[wave][c * 64 + 16 * j + 8 * h] = o;
        }
        __asm__ volatile("s_wait_dscnt 0" ::: "memory");

        // P^T as B-operand (lane = query col, contiguous keys in LDS row)
        bf16x16 bp[2];
#pragma unroll
        for (int ss = 0; ss < 2; ++ss)
            bp[ss] = load_frag_b(&lds[wave][32 * ss], 64, lane);

        // O^T += Vt_tile @ P^T  (Vt rows are contiguous in key index -> A loads)
#pragma unroll
        for (int j2 = 0; j2 < 4; ++j2)
#pragma unroll
            for (int ss = 0; ss < 2; ++ss) {
                bf16x16 av = load_frag_a(Vt + (size_t)(hh * HD + 16 * j2) * NTOK + n0 + 32 * ss,
                                         NTOK, lane);
                oacc[j2] = WMMA_BF16(av, bp[ss], oacc[j2]);
            }
    }
    // epilogue: divide by per-lane l; contiguous 16B stores (lane = token row)
    const float inv = 1.0f / lrun;
    __bf16* orow = O + (size_t)(m0 + c) * TDIM + hh * HD;
#pragma unroll
    for (int j2 = 0; j2 < 4; ++j2) {
        bf16x8 o;
#pragma unroll
        for (int v = 0; v < 8; ++v) o[v] = f2bf(oacc[j2][v] * inv);
        *(bf16x8*)(orow + 16 * j2 + 8 * h) = o;
    }
}

// ---------- host launcher -----------------------------------------------------
extern "C" void kernel_launch(void* const* d_in, const int* in_sizes, int n_in,
                              void* d_out, int out_size, void* d_ws, size_t ws_size,
                              hipStream_t stream) {
    const float* x  = (const float*)d_in[0];
    const float* Wq = (const float*)d_in[1];
    const float* bq = (const float*)d_in[2];
    const float* Wk = (const float*)d_in[3];
    const float* bk = (const float*)d_in[4];
    const float* Wv = (const float*)d_in[5];
    const float* bv = (const float*)d_in[6];
    const float* Wo = (const float*)d_in[7];
    const float* bo = (const float*)d_in[8];

    __bf16* ws  = (__bf16*)d_ws;
    __bf16* xb  = ws;                                   // [NTOK, TDIM]
    __bf16* Wqb = xb  + (size_t)NTOK * TDIM;            // [TDIM, TDIM]
    __bf16* Wkb = Wqb + (size_t)TDIM * TDIM;
    __bf16* Wvb = Wkb + (size_t)TDIM * TDIM;
    __bf16* Wob = Wvb + (size_t)TDIM * TDIM;
    __bf16* Qb  = Wob + (size_t)TDIM * TDIM;            // [NTOK, TDIM]
    __bf16* Kb  = Qb  + (size_t)NTOK * TDIM;            // [NTOK, TDIM]
    __bf16* Vtb = Kb  + (size_t)NTOK * TDIM;            // [TDIM, NTOK] (V^T)
    __bf16* Ob  = Vtb + (size_t)NTOK * TDIM;            // [NTOK, TDIM]

    cvt_f32_bf16<<<2048, 256, 0, stream>>>(x, xb, NTOK * TDIM);
    cvt_f32_bf16<<<1024, 256, 0, stream>>>(Wq, Wqb, TDIM * TDIM);
    cvt_f32_bf16<<<1024, 256, 0, stream>>>(Wk, Wkb, TDIM * TDIM);
    cvt_f32_bf16<<<1024, 256, 0, stream>>>(Wv, Wvb, TDIM * TDIM);
    cvt_f32_bf16<<<1024, 256, 0, stream>>>(Wo, Wob, TDIM * TDIM);

    dim3 g(NTOK / 128, TDIM / 256);
    // Q = (x Wq^T + bq) * (1/hd)   (folds reference's net q scale)
    gemm_wmma<0><<<g, 256, 0, stream>>>(xb, Wqb, bq, Qb, NTOK, TDIM, TDIM, 0, 1.0f / HD);
    gemm_wmma<0><<<g, 256, 0, stream>>>(xb, Wkb, bk, Kb, NTOK, TDIM, TDIM, 0, 1.0f);
    // V stored transposed: Vt[dim, token]
    gemm_wmma<1><<<g, 256, 0, stream>>>(xb, Wvb, bv, Vtb, NTOK, TDIM, TDIM, NTOK, 1.0f);

    attn_flash<<<512, 256, 0, stream>>>(Qb, Kb, Vtb, Ob);

    // out = Ob Wo^T + bo  (fp32)
    gemm_wmma<2><<<g, 256, 0, stream>>>(Ob, Wob, bo, d_out, NTOK, TDIM, TDIM, 0, 1.0f);
}
